// ConvEncoder_61727269978300
// MI455X (gfx1250) — 8-bit matrix operand order repaired and checked by reading, compiled, not run on hardware
//
#include <hip/hip_runtime.h>


typedef int v8i __attribute__((ext_vector_type(8)));

static constexpr int K      = 2048;
static constexpr int OSTR   = 2 * K;
static constexpr int TK     = 56;
static constexpr int NTILES = (K + TK - 1) / TK;
static constexpr int GX     = 16;

__global__ __launch_bounds__(32)
void conv_enc_wmma_kernel(const float* __restrict__ in, float* __restrict__ out)
{
    const int lane = threadIdx.x;
    const int rlo  = lane & 15;
    const int hi   = lane >> 4;
    const int ncol = rlo;
    const int p    = ncol & 1;
    const int ttl  = ncol >> 1;

    v8i Bm[7];
#pragma unroll
    for (int m = 0; m < 7; ++m) {
        const int tt  = 8 * m + ttl;
        const int k_u = tt;
        const int k_1 = (tt >= 1) ? tt - 1 : 62;
        const int k_2 = (tt >= 2) ? tt - 2 : ((tt == 1) ? 62 : 63);
        v8i breg;
#pragma unroll
        for (int v = 0; v < 8; ++v) {
            const int kbase = 16 * (v >> 1) + 8 * hi + 4 * (v & 1);
            unsigned w = 0u;
#pragma unroll
            for (int i = 0; i < 4; ++i) {
                const int k = kbase + i;
                const unsigned bit =
                    (unsigned)((k == k_u) | (k == k_2) |
                               ((p != 0) & (k == k_1)));
                w |= bit << (8 * i);
            }
            breg[v] = (int)w;
        }
        Bm[m] = breg;
    }

    const float* rowp  = in + (size_t)(blockIdx.y * 16 + rlo) * K;
    const int    drow0 = blockIdx.y * 16 + 8 * hi;

    for (int tile = blockIdx.x; tile < NTILES; tile += GX) {
        const int t0 = tile * TK;

        if (t0 + TK * GX < K) __builtin_prefetch(rowp + t0 + TK * GX, 0, 0);

        v8i A;
        const bool fast = (t0 >= 2) && (t0 + 62 <= K);
        if (fast) {
            const float* base = rowp + t0 + 8 * hi;
            float4 q[8];
#pragma unroll
            for (int v = 0; v < 8; ++v) {
                const int koff = 16 * (v >> 1) + 4 * (v & 1);
                q[v] = *(const float4*)(base + koff);
            }
            const float c1 = rowp[t0 - 1];
            const float c2 = rowp[t0 - 2];
#pragma unroll
            for (int v = 0; v < 8; ++v) {
                float b0 = q[v].x, b1 = q[v].y, b2 = q[v].z, b3 = q[v].w;
                if (v == 7) {
                    b2 = hi ? c1 : b2;
                    b3 = hi ? c2 : b3;
                }
                unsigned w = (b0 != 0.0f) ? 1u : 0u;
                w |= (b1 != 0.0f) ? (1u << 8)  : 0u;
                w |= (b2 != 0.0f) ? (1u << 16) : 0u;
                w |= (b3 != 0.0f) ? (1u << 24) : 0u;
                A[v] = (int)w;
            }
        } else {
            float fv[8][4];
#pragma unroll
            for (int v = 0; v < 8; ++v) {
                const int kstart = 16 * (v >> 1) + 8 * hi + 4 * (v & 1);
#pragma unroll
                for (int i = 0; i < 4; ++i) {
                    const int k   = kstart + i;
                    int col = (k >= 62) ? (t0 - (k - 61)) : (t0 + k);
                    int colc = col < 0 ? 0 : (col >= K ? K - 1 : col);
                    fv[v][i] = rowp[colc];
                }
            }
#pragma unroll
            for (int v = 0; v < 8; ++v) {
                const int kstart = 16 * (v >> 1) + 8 * hi + 4 * (v & 1);
                unsigned w = 0u;
#pragma unroll
                for (int i = 0; i < 4; ++i) {
                    const int k   = kstart + i;
                    const int col = (k >= 62) ? (t0 - (k - 61)) : (t0 + k);
                    const unsigned bit =
                        ((fv[v][i] != 0.0f) && (col >= 0)) ? 1u : 0u;
                    w |= bit << (8 * i);
                }
                A[v] = (int)w;
            }
        }

        v8i Dv[7];
#pragma unroll
        for (int m = 0; m < 7; ++m) {
            v8i C = {0, 0, 0, 0, 0, 0, 0, 0};
            Dv[m] = __builtin_amdgcn_wmma_i32_16x16x64_iu8(
                false, A, false, Bm[m], C, false, false);
        }

        float* op = out + (size_t)drow0 * OSTR + 2 * (size_t)t0 + ncol;
        if (fast) {
#pragma unroll
            for (int m = 0; m < 7; ++m)
#pragma unroll
                for (int r = 0; r < 8; ++r)
                    op[r * OSTR + 16 * m] = (float)(Dv[m][r] & 1);
        } else {
#pragma unroll
            for (int m = 0; m < 7; ++m) {
                const int t = t0 + 8 * m + ttl;
                if (t < K) {
#pragma unroll
                    for (int r = 0; r < 8; ++r)
                        op[r * OSTR + 16 * m] = (float)(Dv[m][r] & 1);
                }
            }
        }
    }
}

extern "C" void kernel_launch(void* const* d_in, const int* in_sizes, int n_in,
                              void* d_out, int out_size, void* d_ws, size_t ws_size,
                              hipStream_t stream) {
    (void)n_in; (void)d_ws; (void)ws_size; (void)out_size;
    const float* in  = (const float*)d_in[0];
    float*       out = (float*)d_out;
    const int B = in_sizes[0] / K;
    dim3 grid(GX, B / 16);
    conv_enc_wmma_kernel<<<grid, 32, 0, stream>>>(in, out);
}
